// SwitchMoE_73117523247425
// MI455X (gfx1250) — matrix operand order repaired and checked by reading, not compiled, not run on hardware
//
#include <hip/hip_runtime.h>
#include <math.h>

#define N_EXP   8
#define CAP     1280
#define CDIM    1024
#define FDIM    4096
#define BTOK    8192   // 4 * 2048

// trivial ext-vector types (trivially-copyable, safe in unions, unlike HIP's uint4)
typedef __attribute__((ext_vector_type(16))) __bf16        v16bf;
typedef __attribute__((ext_vector_type(8)))  float         v8f;
typedef __attribute__((ext_vector_type(4))) unsigned int   u32x4;
typedef __attribute__((ext_vector_type(2))) unsigned int   u32x2;
typedef __attribute__((ext_vector_type(4)))  float         f32x4;
typedef __attribute__((ext_vector_type(8)))  int           i32x8;
typedef __attribute__((ext_vector_type(4)))  int           i32x4;

__device__ __forceinline__ unsigned short f2bf(float f) {
    unsigned int u = __float_as_uint(f);
    unsigned int r = u + 0x7fffu + ((u >> 16) & 1u);   // round-to-nearest-even
    return (unsigned short)(r >> 16);
}

union BfFrag { u32x4 u[2]; v16bf v; };

__device__ __forceinline__ v8f wmma_bf(u32x4 a0, u32x4 a1, u32x4 b0, u32x4 b1, v8f c) {
    BfFrag a, b;
    a.u[0] = a0; a.u[1] = a1;
    b.u[0] = b0; b.u[1] = b1;
    return __builtin_amdgcn_wmma_f32_16x16x32_bf16(
        /*neg_a=*/false, a.v, /*neg_b=*/false, b.v,
        /*c_mod=*/(short)0, c, /*reuse_a=*/false, /*reuse_b=*/false);
}

// -------------------------------------------------------------- TDM 2-D tile load
// D# per CDNA5 ISA ch.8: group0 = {count/flags, lds_addr, global_addr lo,
// global_addr hi | type=2}; group1 packs data_size=2B, LDS pad (64B rows +16B pad:
// interval code 3 = 16 DWORDs, amount code 3 = 4 DWORDs), tensor dims (K x rows),
// tile dims (32 x tile_rows), dim0 stride = K.  2-D tensor -> groups 2/3 unused
// (this toolchain's 6-arg builtin: g0, g1, g2, g3, g4, cpol -- zero-fill extras).
__device__ __forceinline__ void tdm_load_2d(unsigned lds_off, const unsigned short* g,
                                            unsigned kdim, unsigned rows,
                                            unsigned tile_rows) {
    unsigned long long ga = (unsigned long long)(size_t)g;
    u32x4 g0;
    g0[0] = 1u;                                           // count=1 (valid), user desc
    g0[1] = lds_off;                                      // lds_addr (bytes)
    g0[2] = (unsigned)ga;                                 // global_addr[31:0]
    g0[3] = ((unsigned)(ga >> 32) & 0x01ffffffu) | (2u << 30);  // ga[56:32] | type=2
    i32x8 g1;
    g1[0] = (int)((1u << 16) | (1u << 20) | (3u << 22) | (3u << 25)); // 2B, pad on
    g1[1] = (int)((kdim & 0xffffu) << 16);                            // tensor_dim0 lo
    g1[2] = (int)(((kdim >> 16) & 0xffffu) | ((rows & 0xffffu) << 16));
    g1[3] = (int)(((rows >> 16) & 0xffffu) | (32u << 16));            // tile_dim0=32
    g1[4] = (int)(tile_rows & 0xffffu);                               // tile_dim1
    g1[5] = (int)kdim;                                                // dim0 stride lo
    g1[6] = 0;
    g1[7] = 0;
    i32x4 z4; z4[0] = 0; z4[1] = 0; z4[2] = 0; z4[3] = 0;
    i32x8 z8;
#pragma unroll
    for (int q = 0; q < 8; ++q) z8[q] = 0;
    __builtin_amdgcn_tensor_load_to_lds(g0, g1, z4, z4, z8, 0);
}

// ---------------------------------------------------------------- fp32 -> bf16
__global__ __launch_bounds__(256) void cvt_kernel(const float* __restrict__ src,
                                                  unsigned short* __restrict__ dst,
                                                  long n4) {
    long i = (long)blockIdx.x * blockDim.x + threadIdx.x;
    if (i >= n4) return;
    f32x4 v = ((const f32x4*)src)[i];
    union { unsigned short s[4]; u32x2 u; } o;
    o.s[0] = f2bf(v.x); o.s[1] = f2bf(v.y); o.s[2] = f2bf(v.z); o.s[3] = f2bf(v.w);
    ((u32x2*)dst)[i] = o.u;
}

// ---------------------------------------------------------------- routing + loss stats
// acc[0]=pen_a sum, acc[1]=lse^2 sum, acc[2..9]=pm(temp) sum, acc[10..17]=P sum,
// acc[18..25]=importance sum, ((int*)acc)[26..33]=expert counts
__global__ __launch_bounds__(256) void gate_kernel(const float* __restrict__ x,
                                                   const float* __restrict__ gw,
                                                   const float* __restrict__ gb,
                                                   float* __restrict__ acc,
                                                   int*   __restrict__ t1i,
                                                   float* __restrict__ t1w) {
    const int wid  = threadIdx.x >> 5;
    const int lane = threadIdx.x & 31;
    const int t    = blockIdx.x * 8 + wid;

    const float* xr = x + (size_t)t * CDIM;
    float xv[32];
#pragma unroll
    for (int k = 0; k < 32; ++k) xv[k] = xr[lane + 32 * k];

    float l[N_EXP];
#pragma unroll
    for (int e = 0; e < N_EXP; ++e) {
        const float* w = gw + e * CDIM;
        float s = 0.f;
#pragma unroll
        for (int k = 0; k < 32; ++k) s += xv[k] * w[lane + 32 * k];
#pragma unroll
        for (int off = 16; off > 0; off >>= 1) s += __shfl_xor(s, off, 32);
        l[e] = s + gb[e];
    }

    if (lane == 0) {
        float mx = l[0]; int bi = 0;
        for (int e = 1; e < N_EXP; ++e) if (l[e] > mx) { mx = l[e]; bi = e; }
        float p[N_EXP], se = 0.f;
        for (int e = 0; e < N_EXP; ++e) { p[e] = expf(l[e] - mx); se += p[e]; }
        float inv = 1.f / se;
        for (int e = 0; e < N_EXP; ++e) p[e] *= inv;
        float lse = mx + logf(se);

        float q[N_EXP], mx2 = -1e30f, se2 = 0.f;
        for (int e = 0; e < N_EXP; ++e) { q[e] = l[e] * (1.0f / 1.66f); mx2 = fmaxf(mx2, q[e]); }
        for (int e = 0; e < N_EXP; ++e) { q[e] = expf(q[e] - mx2); se2 += q[e]; }
        float inv2 = 1.f / se2, pen = 0.f;
        for (int e = 0; e < N_EXP; ++e) { q[e] *= inv2; pen += q[e] * (1.f - q[e]); }

        atomicAdd(&acc[0], pen);
        atomicAdd(&acc[1], lse * lse);
        for (int e = 0; e < N_EXP; ++e) {
            atomicAdd(&acc[2 + e],  q[e]);
            atomicAdd(&acc[10 + e], p[e]);
        }
        atomicAdd(&acc[18 + bi], p[bi]);
        atomicAdd((int*)acc + 26 + bi, 1);
        t1i[t] = bi;
        t1w[t] = p[bi];
    }
}

// ---------------------------------------------------------------- order-preserving scan
__global__ __launch_bounds__(256) void scan_kernel(const int* __restrict__ idx,
                                                   int* __restrict__ pos) {
    __shared__ int sh[256];
    __shared__ int base[N_EXP];
    const int tid = threadIdx.x;
    if (tid < N_EXP) base[tid] = 0;
    __syncthreads();
    for (int c = 0; c < BTOK / 256; ++c) {
        int t = c * 256 + tid;
        int e = idx[t];
        sh[tid] = e;
        __syncthreads();
        int rank = 0;
        for (int j = 0; j < tid; ++j) rank += (sh[j] == e);
        pos[t] = base[e] + rank;
        __syncthreads();
        if (tid < N_EXP) {
            int cnt = 0;
            for (int j = 0; j < 256; ++j) cnt += (sh[j] == tid);
            base[tid] += cnt;
        }
        __syncthreads();
    }
}

__global__ __launch_bounds__(256) void init_invmap(int* __restrict__ invmap) {
    int i = blockIdx.x * blockDim.x + threadIdx.x;
    if (i < N_EXP * CAP) invmap[i] = -1;
}

// ---------------------------------------------------------------- dispatch (fp32 -> bf16 buf)
__global__ __launch_bounds__(256) void dispatch_kernel(const float* __restrict__ x,
                                                       const int* __restrict__ idx,
                                                       const int* __restrict__ pos,
                                                       unsigned short* __restrict__ buf,
                                                       int* __restrict__ invmap) {
    const int t  = blockIdx.x;
    const int ps = pos[t];
    if (ps >= CAP) return;                 // dropped token (uniform per block)
    const int e = idx[t];
    if (threadIdx.x == 0) invmap[e * CAP + ps] = t;
    f32x4 v = *(const f32x4*)(x + (size_t)t * CDIM + threadIdx.x * 4);
    union { unsigned short s[4]; u32x2 u; } o;
    o.s[0] = f2bf(v.x); o.s[1] = f2bf(v.y); o.s[2] = f2bf(v.z); o.s[3] = f2bf(v.w);
    *(u32x2*)&buf[((size_t)e * CAP + ps) * CDIM + threadIdx.x * 4] = o.u;
}

// ---------------------------------------------------------------- GEMM1 + GELU
// h[e,m,n] = gelu( sum_k buf[e,m,k]*w1[e,n,k] + b1[e,n] ),  M=CAP N=FDIM K=CDIM
// Block tile 64(M)x128(N); 4 waves, each 32x64 (2x4 wmma accums).
// A/B K-tiles DMA'd into double-buffered LDS by the TDM; wave 0 drives it.
#define LDS_STRIDE 40   // 32 + 8 bf16 pad (TDM pad_interval=3, pad_amount=3)
__global__ __launch_bounds__(128) void ffn1_kernel(const unsigned short* __restrict__ bufp,
                                                   const unsigned short* __restrict__ w1b,
                                                   const float* __restrict__ b1,
                                                   unsigned short* __restrict__ hp) {
    const int e  = blockIdx.z;
    const int n0 = blockIdx.x * 128;
    const int m0 = blockIdx.y * 64;
    const unsigned short* A  = bufp + (size_t)e * CAP  * CDIM + (size_t)m0 * CDIM;
    const unsigned short* Bm = w1b  + (size_t)e * FDIM * CDIM + (size_t)n0 * CDIM;
    const float* bias        = b1   + (size_t)e * FDIM;

    __shared__ unsigned short sA[2][64  * LDS_STRIDE];
    __shared__ unsigned short sB[2][128 * LDS_STRIDE];

    const int tid  = threadIdx.x;
    const int lane = tid & 31;
    const int wid  = tid >> 5;
    const int wm   = (wid >> 1) * 32;
    const int wn   = (wid & 1) * 64;

    v8f zero;
#pragma unroll
    for (int q = 0; q < 8; ++q) zero[q] = 0.f;
    v8f accv[2][4];
#pragma unroll
    for (int i = 0; i < 2; ++i)
#pragma unroll
        for (int j = 0; j < 4; ++j) accv[i][j] = zero;

    const int frow  = lane & 15;
    const int akoff = (lane >> 4) * 8;    // A frag: lanes 16-31 hold K+8
    const int bkoff = (lane >> 4) * 8;    // B frag: lanes 16-31 hold K+8

    const unsigned ldsA0 = (unsigned)(size_t)&sA[0][0];
    const unsigned ldsA1 = (unsigned)(size_t)&sA[1][0];
    const unsigned ldsB0 = (unsigned)(size_t)&sB[0][0];
    const unsigned ldsB1 = (unsigned)(size_t)&sB[1][0];

    const int NT = CDIM / 32;
    if (wid == 0) {                                   // prologue: tile 0 -> buffer 0
        tdm_load_2d(ldsA0, A,  CDIM, CAP,  64);
        tdm_load_2d(ldsB0, Bm, CDIM, FDIM, 128);
    }

    for (int kt = 0; kt < NT; ++kt) {
        if (wid == 0) {
            if (kt + 1 < NT) {                        // issue next tile, keep 2 in flight
                const int nb = (kt + 1) & 1;
                tdm_load_2d(nb ? ldsA1 : ldsA0, A  + (kt + 1) * 32, CDIM, CAP,  64);
                tdm_load_2d(nb ? ldsB1 : ldsB0, Bm + (kt + 1) * 32, CDIM, FDIM, 128);
                __builtin_amdgcn_s_wait_tensorcnt(2); // current tile complete
            } else {
                __builtin_amdgcn_s_wait_tensorcnt(0);
            }
        }
        __syncthreads();
        asm volatile("" ::: "memory");                // LDS written by DMA, not by us

        const unsigned short* sAb = sA[kt & 1];
        const unsigned short* sBb = sB[kt & 1];
        u32x4 a0[2], a1[2], b0[4], b1v[4];
#pragma unroll
        for (int i = 0; i < 2; ++i) {
            int rr = wm + i * 16 + frow;
            a0[i] = *(const u32x4*)&sAb[rr * LDS_STRIDE + akoff];
            a1[i] = *(const u32x4*)&sAb[rr * LDS_STRIDE + akoff + 16];
        }
#pragma unroll
        for (int j = 0; j < 4; ++j) {
            int rr = wn + j * 16 + frow;
            b0[j]  = *(const u32x4*)&sBb[rr * LDS_STRIDE + bkoff];
            b1v[j] = *(const u32x4*)&sBb[rr * LDS_STRIDE + bkoff + 16];
        }
#pragma unroll
        for (int i = 0; i < 2; ++i)
#pragma unroll
            for (int j = 0; j < 4; ++j)
                accv[i][j] = wmma_bf(a0[i], a1[i], b0[j], b1v[j], accv[i][j]);
        __syncthreads();                              // readers done before DMA reuses buffer
    }

    const int rowsel = 8 * (lane >> 4);
    const int colsel = lane & 15;
#pragma unroll
    for (int i = 0; i < 2; ++i) {
#pragma unroll
        for (int j = 0; j < 4; ++j) {
            int col  = n0 + wn + j * 16 + colsel;
            float bv = bias[col];
#pragma unroll
            for (int r = 0; r < 8; ++r) {
                int row = m0 + wm + i * 16 + rowsel + r;
                float v = accv[i][j][r] + bv;
                float g = 0.5f * v * (1.0f + tanhf(0.7978845608028654f * (v + 0.044715f * v * v * v)));
                hp[(size_t)e * CAP * FDIM + (size_t)row * FDIM + col] = f2bf(g);
            }
        }
    }
}

// ---------------------------------------------------------------- GEMM2 + combine/scatter
// out[t,n] = top1w[t] * ( sum_k h[e,m,k]*w2[e,n,k] + b2[e,n] ),  t = invmap[e,m]
__global__ __launch_bounds__(128) void ffn2_kernel(const unsigned short* __restrict__ hp,
                                                   const unsigned short* __restrict__ w2b,
                                                   const float* __restrict__ b2,
                                                   const int*   __restrict__ invmap,
                                                   const float* __restrict__ t1w,
                                                   float* __restrict__ outp) {
    const int e  = blockIdx.z;
    const int n0 = blockIdx.x * 128;
    const int m0 = blockIdx.y * 64;
    const unsigned short* A  = hp  + (size_t)e * CAP  * FDIM + (size_t)m0 * FDIM;
    const unsigned short* Bm = w2b + (size_t)e * CDIM * FDIM + (size_t)n0 * FDIM;
    const float* bias        = b2  + (size_t)e * CDIM;

    __shared__ unsigned short sA[2][64  * LDS_STRIDE];
    __shared__ unsigned short sB[2][128 * LDS_STRIDE];

    const int tid  = threadIdx.x;
    const int lane = tid & 31;
    const int wid  = tid >> 5;
    const int wm   = (wid >> 1) * 32;
    const int wn   = (wid & 1) * 64;

    v8f zero;
#pragma unroll
    for (int q = 0; q < 8; ++q) zero[q] = 0.f;
    v8f accv[2][4];
#pragma unroll
    for (int i = 0; i < 2; ++i)
#pragma unroll
        for (int j = 0; j < 4; ++j) accv[i][j] = zero;

    const int frow  = lane & 15;
    const int akoff = (lane >> 4) * 8;
    const int bkoff = (lane >> 4) * 8;

    const unsigned ldsA0 = (unsigned)(size_t)&sA[0][0];
    const unsigned ldsA1 = (unsigned)(size_t)&sA[1][0];
    const unsigned ldsB0 = (unsigned)(size_t)&sB[0][0];
    const unsigned ldsB1 = (unsigned)(size_t)&sB[1][0];

    const int NT = FDIM / 32;
    if (wid == 0) {
        tdm_load_2d(ldsA0, A,  FDIM, CAP,  64);
        tdm_load_2d(ldsB0, Bm, FDIM, CDIM, 128);
    }

    for (int kt = 0; kt < NT; ++kt) {
        if (wid == 0) {
            if (kt + 1 < NT) {
                const int nb = (kt + 1) & 1;
                tdm_load_2d(nb ? ldsA1 : ldsA0, A  + (kt + 1) * 32, FDIM, CAP,  64);
                tdm_load_2d(nb ? ldsB1 : ldsB0, Bm + (kt + 1) * 32, FDIM, CDIM, 128);
                __builtin_amdgcn_s_wait_tensorcnt(2);
            } else {
                __builtin_amdgcn_s_wait_tensorcnt(0);
            }
        }
        __syncthreads();
        asm volatile("" ::: "memory");

        const unsigned short* sAb = sA[kt & 1];
        const unsigned short* sBb = sB[kt & 1];
        u32x4 a0[2], a1[2], b0[4], b1v[4];
#pragma unroll
        for (int i = 0; i < 2; ++i) {
            int rr = wm + i * 16 + frow;
            a0[i] = *(const u32x4*)&sAb[rr * LDS_STRIDE + akoff];
            a1[i] = *(const u32x4*)&sAb[rr * LDS_STRIDE + akoff + 16];
        }
#pragma unroll
        for (int j = 0; j < 4; ++j) {
            int rr = wn + j * 16 + frow;
            b0[j]  = *(const u32x4*)&sBb[rr * LDS_STRIDE + bkoff];
            b1v[j] = *(const u32x4*)&sBb[rr * LDS_STRIDE + bkoff + 16];
        }
#pragma unroll
        for (int i = 0; i < 2; ++i)
#pragma unroll
            for (int j = 0; j < 4; ++j)
                accv[i][j] = wmma_bf(a0[i], a1[i], b0[j], b1v[j], accv[i][j]);
        __syncthreads();
    }

    const int rowsel = 8 * (lane >> 4);
    const int colsel = lane & 15;
#pragma unroll
    for (int i = 0; i < 2; ++i) {
#pragma unroll
        for (int r = 0; r < 8; ++r) {
            int row = m0 + wm + i * 16 + rowsel + r;
            int t   = invmap[e * CAP + row];
            if (t < 0) continue;
            float w = t1w[t];
#pragma unroll
            for (int j = 0; j < 4; ++j) {
                int col = n0 + wn + j * 16 + colsel;
                outp[(size_t)t * CDIM + col] = w * (accv[i][j][r] + bias[col]);
            }
        }
    }
}

// ---------------------------------------------------------------- finalize aux loss
__global__ void loss_kernel(const float* __restrict__ acc, float* __restrict__ outp) {
    if (threadIdx.x != 0 || blockIdx.x != 0) return;
    const int* cnt = (const int*)(acc + 26);
    float pen_a = acc[0] / (float)(BTOK * N_EXP);
    float penb = 0.f, load = 0.f, imp[N_EXP], isum = 0.f;
    for (int e = 0; e < N_EXP; ++e) {
        float pm = acc[2 + e] / (float)BTOK;
        penb += pm * (1.f - pm);
        float Pi = acc[10 + e] / (float)BTOK;
        float fi = (float)cnt[e] / (float)BTOK;
        load += fi * Pi;
        imp[e] = acc[18 + e]; isum += imp[e];
    }
    float pen_b = 1.f / (float)N_EXP - penb / (float)N_EXP;
    float z     = 0.001f * (acc[1] / (float)BTOK);
    float loadl = 0.01f * (float)N_EXP * load;
    float m = isum / (float)N_EXP, var = 0.f;
    for (int e = 0; e < N_EXP; ++e) { float d = imp[e] - m; var += d * d; }
    var /= (float)N_EXP;
    outp[0] = 0.01f * (pen_a + pen_b) + z + loadl + 0.01f * (var / (m * m));
}

// ---------------------------------------------------------------- host launcher
extern "C" void kernel_launch(void* const* d_in, const int* in_sizes, int n_in,
                              void* d_out, int out_size, void* d_ws, size_t ws_size,
                              hipStream_t stream) {
    (void)in_sizes; (void)n_in; (void)ws_size;
    const float* x  = (const float*)d_in[0];
    const float* gw = (const float*)d_in[1];
    const float* gb = (const float*)d_in[2];
    const float* w1 = (const float*)d_in[3];
    const float* b1 = (const float*)d_in[4];
    const float* w2 = (const float*)d_in[5];
    const float* b2 = (const float*)d_in[6];
    float* out = (float*)d_out;

    char* ws = (char*)d_ws;
    float*          accp   = (float*)ws;                                  //   256 B
    int*            t1i    = (int*)(ws + 256);                            //  32 KB
    float*          t1w    = (float*)(ws + 33024);                        //  32 KB
    int*            pos    = (int*)(ws + 65792);                          //  32 KB
    int*            invmap = (int*)(ws + 98560);                          //  40 KB
    unsigned short* buf    = (unsigned short*)(ws + 139520);              //  20 MB
    unsigned short* hbuf   = (unsigned short*)(ws + 21111040);            //  80 MB
    unsigned short* w1b    = (unsigned short*)(ws + 104997120);           //  64 MB
    unsigned short* w2b    = (unsigned short*)(ws + 172105984);           //  64 MB

    (void)hipMemsetAsync(d_out, 0, (size_t)out_size * sizeof(float), stream); // dropped tokens -> 0
    (void)hipMemsetAsync(accp, 0, 256, stream);

    const long n4 = (long)N_EXP * FDIM * CDIM / 4;
    cvt_kernel<<<(int)(n4 / 256), 256, 0, stream>>>(w1, w1b, n4);
    cvt_kernel<<<(int)(n4 / 256), 256, 0, stream>>>(w2, w2b, n4);

    gate_kernel<<<BTOK / 8, 256, 0, stream>>>(x, gw, gb, accp, t1i, t1w);
    scan_kernel<<<1, 256, 0, stream>>>(t1i, pos);
    init_invmap<<<(N_EXP * CAP + 255) / 256, 256, 0, stream>>>(invmap);
    dispatch_kernel<<<BTOK, 256, 0, stream>>>(x, t1i, pos, buf, invmap);

    ffn1_kernel<<<dim3(FDIM / 128, CAP / 64, N_EXP), 128, 0, stream>>>(buf, w1b, b1, hbuf);
    ffn2_kernel<<<dim3(CDIM / 128, CAP / 64, N_EXP), 128, 0, stream>>>(hbuf, w2b, b2, invmap, t1w, out);

    loss_kernel<<<1, 1, 0, stream>>>(accp, out + (size_t)BTOK * CDIM);
}
